// HCF_module_69020124447045
// MI455X (gfx1250) — compile-verified, matrix operand order checked by reading, not run on hardware
//
#include <hip/hip_runtime.h>
#include <hip/hip_bf16.h>
#include <math.h>

// ---------------------------------------------------------------------------
// SC2-PCR style registration pipeline for MI455X (gfx1250, wave32).
//   K1: per-seed top-200 of SC2_measure + gather knn points   (LDS argmax loop)
//   K2: corr_filter x4 (200->100->50->25->12)                 (LDS, VALU)
//   K3: 12x12 power iteration + weighted Kabsch (3x3 SVD)     (1 thread/seed)
//   K4: verification: pred = [x y z 1] @ [R^T; t] via v_wmma_f32_16x16x32_f16
//   K5: argmax fitness -> write best 4x4 transform
// ---------------------------------------------------------------------------

#define NSEEDS 512
#define NCH    2048
#define TOPK_  200

typedef __attribute__((ext_vector_type(16))) _Float16 v16h;
typedef __attribute__((ext_vector_type(8)))  float    v8f;

__device__ __forceinline__ float pdist3(const float* a, const float* b) {
  float dx = a[0]-b[0], dy = a[1]-b[1], dz = a[2]-b[2];
  return sqrtf(fmaxf(dx*dx + dy*dy + dz*dz, 1e-12f));
}

// ------------------------- K1: top-200 + gather -----------------------------
__global__ void topk_gather_kernel(const float* __restrict__ sc2m,
                                   const float* __restrict__ srcp,
                                   const float* __restrict__ tgtp,
                                   float* __restrict__ src_out,   // [512][200][3]
                                   float* __restrict__ tgt_out) {
  __shared__ float vals[NCH];
  __shared__ float rv[256];
  __shared__ int   ri[256];
  __shared__ int   selIdx;
  const int s = blockIdx.x;
  const int t = threadIdx.x;
  const float* row = sc2m + (size_t)s * NCH;
  for (int i = t; i < NCH; i += 256) vals[i] = row[i];
  __syncthreads();

  for (int j = 0; j < TOPK_; ++j) {
    float bv = -INFINITY; int bi = 0x7fffffff;
    for (int i = t; i < NCH; i += 256) {
      float v = vals[i];
      if (v > bv) { bv = v; bi = i; }       // strided ascending -> first max kept
    }
    rv[t] = bv; ri[t] = bi;
    __syncthreads();
    for (int o = 128; o > 0; o >>= 1) {
      if (t < o) {
        float ov = rv[t+o]; int oi = ri[t+o];
        if (ov > rv[t] || (ov == rv[t] && oi < ri[t])) { rv[t] = ov; ri[t] = oi; }
      }
      __syncthreads();
    }
    if (t == 0) { selIdx = ri[0]; vals[ri[0]] = -INFINITY; }
    __syncthreads();
    if (t < 6) {
      int p = selIdx;
      if (t < 3) src_out[((size_t)s*TOPK_ + j)*3 + t]     = srcp[(size_t)p*3 + t];
      else       tgt_out[((size_t)s*TOPK_ + j)*3 + (t-3)] = tgtp[(size_t)p*3 + (t-3)];
    }
    __syncthreads();
  }
}

// ------------------------- K2: corr_filter ----------------------------------
__global__ void corr_filter_kernel(const float* __restrict__ src_in,
                                   const float* __restrict__ tgt_in,
                                   float* __restrict__ src_out,
                                   float* __restrict__ tgt_out,
                                   int k, int kf) {
  __shared__ float sp[TOPK_*3];
  __shared__ float tp[TOPK_*3];
  __shared__ float hard0[TOPK_];
  __shared__ float scv[256];
  __shared__ float rv[256];
  __shared__ int   ri[256];
  __shared__ int   selIdx;
  const int s = blockIdx.x;
  const int t = threadIdx.x;
  const float* S = src_in + (size_t)s * k * 3;
  const float* T = tgt_in + (size_t)s * k * 3;
  for (int i = t; i < k*3; i += 256) { sp[i] = S[i]; tp[i] = T[i]; }
  __syncthreads();

  if (t < k) {
    float ds = pdist3(&sp[0], &sp[t*3]);
    float dt = pdist3(&tp[0], &tp[t*3]);
    hard0[t] = (fabsf(ds - dt) < 0.1f) ? 1.f : 0.f;
  }
  __syncthreads();

  for (int j = t; j < k; j += 256) {
    const float sx = sp[j*3], sy = sp[j*3+1], sz = sp[j*3+2];
    const float tx = tp[j*3], ty = tp[j*3+1], tz = tp[j*3+2];
    float acc = 0.f;
    for (int kk = 0; kk < k; ++kk) {
      float ax = sp[kk*3]-sx, ay = sp[kk*3+1]-sy, az = sp[kk*3+2]-sz;
      float ds = sqrtf(fmaxf(ax*ax + ay*ay + az*az, 1e-12f));
      float bx = tp[kk*3]-tx, by = tp[kk*3+1]-ty, bz = tp[kk*3+2]-tz;
      float dt = sqrtf(fmaxf(bx*bx + by*by + bz*bz, 1e-12f));
      acc += hard0[kk] * ((fabsf(ds - dt) < 0.1f) ? 1.f : 0.f);
    }
    scv[j] = acc;
  }
  __syncthreads();

  for (int j = 0; j < kf; ++j) {
    if (t < k) { rv[t] = scv[t]; ri[t] = t; }
    else       { rv[t] = -INFINITY; ri[t] = 0x7fffffff; }
    __syncthreads();
    for (int o = 128; o > 0; o >>= 1) {
      if (t < o) {
        float ov = rv[t+o]; int oi = ri[t+o];
        if (ov > rv[t] || (ov == rv[t] && oi < ri[t])) { rv[t] = ov; ri[t] = oi; }
      }
      __syncthreads();
    }
    if (t == 0) { selIdx = ri[0]; scv[ri[0]] = -INFINITY; }
    __syncthreads();
    if (t < 6) {
      int p = selIdx;
      if (t < 3) src_out[((size_t)s*kf + j)*3 + t]     = sp[p*3 + t];
      else       tgt_out[((size_t)s*kf + j)*3 + (t-3)] = tp[p*3 + (t-3)];
    }
    __syncthreads();
  }
}

// ------------------------- K3: eig + Kabsch ---------------------------------
__device__ __forceinline__ float det3(const float M[3][3]) {
  return M[0][0]*(M[1][1]*M[2][2]-M[1][2]*M[2][1])
       - M[0][1]*(M[1][0]*M[2][2]-M[1][2]*M[2][0])
       + M[0][2]*(M[1][0]*M[2][1]-M[1][1]*M[2][0]);
}

__global__ void solve_kernel(const float* __restrict__ src_knn,
                             const float* __restrict__ tgt_knn,
                             float* __restrict__ Tbuf) {
  const int s = blockIdx.x * blockDim.x + threadIdx.x;
  if (s >= NSEEDS) return;
  const int K = 12;
  float A[12][3], B[12][3];
  const float* sp = src_knn + (size_t)s * K * 3;
  const float* tp = tgt_knn + (size_t)s * K * 3;
  for (int i = 0; i < K; ++i)
    for (int c = 0; c < 3; ++c) { A[i][c] = sp[i*3+c]; B[i][c] = tp[i*3+c]; }

  // compatibility matrix M (diag zeroed)
  float M[12][12];
  for (int i = 0; i < K; ++i) {
    M[i][i] = 0.f;
    for (int j = i+1; j < K; ++j) {
      float ds = pdist3(A[i], A[j]);
      float dt = pdist3(B[i], B[j]);
      float cr = fabsf(ds - dt);
      float v  = fmaxf(1.f - cr*cr*100.f, 0.f);   // D_THRE^2 = 0.01
      M[i][j] = v; M[j][i] = v;
    }
  }
  // power iteration
  float v[12]; for (int i = 0; i < K; ++i) v[i] = 1.f;
  for (int it = 0; it < 10; ++it) {
    float nv[12]; float nrm = 0.f;
    for (int i = 0; i < K; ++i) {
      float a = 0.f;
      for (int j = 0; j < K; ++j) a += M[i][j] * v[j];
      nv[i] = a; nrm += a*a;
    }
    nrm = sqrtf(nrm) + 1e-6f;
    for (int i = 0; i < K; ++i) v[i] = nv[i] / nrm;
  }
  float sv = 0.f; for (int i = 0; i < K; ++i) sv += v[i];
  sv += 1e-6f;
  float w[12]; for (int i = 0; i < K; ++i) w[i] = v[i] / sv;
  float wsum = 0.f; for (int i = 0; i < K; ++i) wsum += w[i];
  wsum += 1e-6f;

  float cA[3] = {0,0,0}, cB[3] = {0,0,0};
  for (int i = 0; i < K; ++i)
    for (int c = 0; c < 3; ++c) { cA[c] += A[i][c]*w[i]; cB[c] += B[i][c]*w[i]; }
  for (int c = 0; c < 3; ++c) { cA[c] /= wsum; cB[c] /= wsum; }

  float H[3][3] = {{0,0,0},{0,0,0},{0,0,0}};
  for (int kk = 0; kk < K; ++kk)
    for (int i = 0; i < 3; ++i)
      for (int j = 0; j < 3; ++j)
        H[i][j] += w[kk] * (A[kk][i]-cA[i]) * (B[kk][j]-cB[j]);

  // SVD of H via Jacobi on S = H^T H
  float S[3][3];
  for (int i = 0; i < 3; ++i)
    for (int j = 0; j < 3; ++j) {
      float a = 0.f;
      for (int m = 0; m < 3; ++m) a += H[m][i]*H[m][j];
      S[i][j] = a;
    }
  float V[3][3] = {{1,0,0},{0,1,0},{0,0,1}};
  const int PQ[3][2] = {{0,1},{0,2},{1,2}};
  for (int sweep = 0; sweep < 12; ++sweep)
    for (int pi = 0; pi < 3; ++pi) {
      int p = PQ[pi][0], q = PQ[pi][1];
      float apq = S[p][q];
      if (fabsf(apq) < 1e-15f) continue;
      float theta = (S[q][q]-S[p][p]) / (2.f*apq);
      float tt = (theta >= 0.f ? 1.f : -1.f) / (fabsf(theta) + sqrtf(theta*theta + 1.f));
      float c = 1.f / sqrtf(tt*tt + 1.f), sn = tt*c;
      float app = S[p][p], aqq = S[q][q];
      S[p][p] = app - tt*apq;
      S[q][q] = aqq + tt*apq;
      S[p][q] = 0.f; S[q][p] = 0.f;
      int r = 3 - p - q;
      float arp = S[r][p], arq = S[r][q];
      S[r][p] = c*arp - sn*arq; S[p][r] = S[r][p];
      S[r][q] = sn*arp + c*arq; S[q][r] = S[r][q];
      for (int m = 0; m < 3; ++m) {
        float vp = V[m][p], vq = V[m][q];
        V[m][p] = c*vp - sn*vq;
        V[m][q] = sn*vp + c*vq;
      }
    }
  float lam[3] = {S[0][0], S[1][1], S[2][2]};
  for (int a = 0; a < 2; ++a)
    for (int b = a+1; b < 3; ++b)
      if (lam[b] > lam[a]) {
        float tl = lam[a]; lam[a] = lam[b]; lam[b] = tl;
        for (int m = 0; m < 3; ++m) { float tv = V[m][a]; V[m][a] = V[m][b]; V[m][b] = tv; }
      }
  float U[3][3];
  for (int c = 0; c < 3; ++c) {
    float sig = sqrtf(fmaxf(lam[c], 0.f));
    float inv = 1.f / fmaxf(sig, 1e-12f);
    for (int m = 0; m < 3; ++m) {
      float hv = 0.f;
      for (int j = 0; j < 3; ++j) hv += H[m][j]*V[j][c];
      U[m][c] = hv * inv;
    }
  }
  float dsign = det3(V) * det3(U);
  float e[3] = {1.f, 1.f, dsign};
  float R[3][3];
  for (int i = 0; i < 3; ++i)
    for (int j = 0; j < 3; ++j) {
      float a = 0.f;
      for (int m = 0; m < 3; ++m) a += V[i][m]*e[m]*U[j][m];
      R[i][j] = a;
    }
  float t3[3];
  for (int i = 0; i < 3; ++i) {
    float a = 0.f;
    for (int j = 0; j < 3; ++j) a += R[i][j]*cA[j];
    t3[i] = cB[i] - a;
  }
  float* To = Tbuf + (size_t)s * 16;
  for (int i = 0; i < 3; ++i) {
    for (int j = 0; j < 3; ++j) To[i*4+j] = R[i][j];
    To[i*4+3] = t3[i];
  }
  To[12] = 0.f; To[13] = 0.f; To[14] = 0.f; To[15] = 1.f;
}

// ------------------------- K4: WMMA verification ----------------------------
// pred(i,c) = [x y z 1] . [R(c,0) R(c,1) R(c,2) t(c)]  -> A(16x32) @ B(32x16)
// via v_wmma_f32_16x16x32_f16; one wave (32 lanes) per seed, 128 tiles of 16 pts.
__global__ void fitness_kernel(const float* __restrict__ srcp,
                               const float* __restrict__ tgtp,
                               const float* __restrict__ Tbuf,
                               float* __restrict__ fit) {
  const int s    = blockIdx.x;
  const int lane = threadIdx.x;            // 0..31 (wave32)
  const int n    = lane & 15;              // output column (coordinate)
  __shared__ float pred[16][3];
  __shared__ int   red[32];

  const float* T = Tbuf + (size_t)s * 16;
  v16h bm = {};
  if (lane < 16 && n < 3) {                // lanes 0..15 carry K=0..3 of B (elements 0..3)
    bm[0] = (_Float16)T[n*4+0];
    bm[1] = (_Float16)T[n*4+1];
    bm[2] = (_Float16)T[n*4+2];
    bm[3] = (_Float16)T[n*4+3];
  }
  int cnt = 0;
  for (int tile = 0; tile < NCH/16; ++tile) {
    v16h am = {};
    if (lane < 16) {                       // lanes 0..15 carry K=0..7 of row M=lane
      const float* p = srcp + (size_t)(tile*16 + lane)*3;
      am[0] = (_Float16)p[0];
      am[1] = (_Float16)p[1];
      am[2] = (_Float16)p[2];
      am[3] = (_Float16)1.0f;
    }
    v8f cm = {};
    v8f dm = __builtin_amdgcn_wmma_f32_16x16x32_f16(
        false, am, false, bm, (short)0, cm, false, false);
    __syncthreads();                        // previous tile fully consumed
    if (n < 3) {                            // D: lanes<16 rows 0..7, lanes>=16 rows 8..15
      int base = (lane < 16) ? 0 : 8;
#pragma unroll
      for (int r = 0; r < 8; ++r) pred[base + r][n] = dm[r];
    }
    __syncthreads();
    if (lane < 16) {
      const float* q = tgtp + (size_t)(tile*16 + lane)*3;
      float dx = pred[lane][0]-q[0], dy = pred[lane][1]-q[1], dz = pred[lane][2]-q[2];
      float d2 = dx*dx + dy*dy + dz*dz;
      cnt += (d2 < 0.01f) ? 1 : 0;          // INLIER_THR^2
    }
  }
  red[lane] = cnt;
  __syncthreads();
  if (lane == 0) {
    int tot = 0;
    for (int i = 0; i < 32; ++i) tot += red[i];
    fit[s] = (float)tot;
  }
}

// ------------------------- K5: select best ----------------------------------
__global__ void select_kernel(const float* __restrict__ fit,
                              const float* __restrict__ Tbuf,
                              float* __restrict__ out) {
  __shared__ float rv[NSEEDS];
  __shared__ int   ri[NSEEDS];
  const int t = threadIdx.x;
  rv[t] = fit[t]; ri[t] = t;
  __syncthreads();
  for (int o = NSEEDS/2; o > 0; o >>= 1) {
    if (t < o) {
      float ov = rv[t+o]; int oi = ri[t+o];
      if (ov > rv[t] || (ov == rv[t] && oi < ri[t])) { rv[t] = ov; ri[t] = oi; }
    }
    __syncthreads();
  }
  if (t < 16) out[t] = Tbuf[(size_t)ri[0]*16 + t];
}

// ---------------------------------------------------------------------------
extern "C" void kernel_launch(void* const* d_in, const int* in_sizes, int n_in,
                              void* d_out, int out_size, void* d_ws, size_t ws_size,
                              hipStream_t stream) {
  const float* sc2m = (const float*)d_in[0];   // (1,512,2048) f32
  const float* srcp = (const float*)d_in[1];   // (1,2048,3)  f32
  const float* tgtp = (const float*)d_in[2];   // (1,2048,3)  f32
  float* out = (float*)d_out;                  // (4,4) f32
  float* ws  = (float*)d_ws;

  const size_t knnSz = (size_t)NSEEDS * TOPK_ * 3;
  float* srcA = ws;
  float* tgtA = srcA + knnSz;
  float* srcB = tgtA + knnSz;
  float* tgtB = srcB + knnSz;
  float* Tbuf = tgtB + knnSz;                  // 512*16
  float* fit  = Tbuf + (size_t)NSEEDS * 16;    // 512

  topk_gather_kernel<<<NSEEDS, 256, 0, stream>>>(sc2m, srcp, tgtp, srcA, tgtA);
  corr_filter_kernel<<<NSEEDS, 256, 0, stream>>>(srcA, tgtA, srcB, tgtB, 200, 100);
  corr_filter_kernel<<<NSEEDS, 256, 0, stream>>>(srcB, tgtB, srcA, tgtA, 100, 50);
  corr_filter_kernel<<<NSEEDS, 256, 0, stream>>>(srcA, tgtA, srcB, tgtB, 50, 25);
  corr_filter_kernel<<<NSEEDS, 256, 0, stream>>>(srcB, tgtB, srcA, tgtA, 25, 12);
  solve_kernel<<<2, 256, 0, stream>>>(srcA, tgtA, Tbuf);
  fitness_kernel<<<NSEEDS, 32, 0, stream>>>(srcp, tgtp, Tbuf, fit);
  select_kernel<<<1, NSEEDS, 0, stream>>>(fit, Tbuf, out);
}
